// MixedScore_MultiHeadAttention_40183714021845
// MI455X (gfx1250) — matrix operand order repaired and checked by reading, not compiled, not run on hardware
//
#include <hip/hip_runtime.h>

// Problem constants (match reference)
#define B_  4
#define L_  512
#define D_  256
#define H_  8
#define DK_ 32
#define MS_ 16
#define SCALE_ 0.17677669529663689f   // DK^-0.5

typedef __attribute__((ext_vector_type(16))) _Float16 v16h;
typedef __attribute__((ext_vector_type(8)))  _Float16 v8h;
typedef __attribute__((ext_vector_type(2)))  _Float16 h2;
typedef __attribute__((ext_vector_type(8)))  float    v8f;

static __device__ __forceinline__ v8f wmma16(v16h a, v16h b, v8f c) {
    // D = A(16x32 f16) x B(32x16 f16) + C(16x16 f32)
    return __builtin_amdgcn_wmma_f32_16x16x32_f16(false, a, false, b, (short)0, c,
                                                  false, false);
}

static __device__ __forceinline__ v16h combine(v8h lo, v8h hi) {
    v16h r;
    #pragma unroll
    for (int e = 0; e < 8; ++e) { r[e] = lo[e]; r[8 + e] = hi[e]; }
    return r;
}

// ---------------------------------------------------------------------------
// Pre-pack A: queries [B*L, D] f32 -> f16 A-fragment order.
//   apack[((mt*8 + kc)*32 + lane)*16 + e]
// grid: (128, 8), block 32
// ---------------------------------------------------------------------------
__global__ __launch_bounds__(32)
void pack_a_kernel(const float* __restrict__ X, _Float16* __restrict__ apack)
{
    const int mt = blockIdx.x, kc = blockIdx.y;
    const int lane = threadIdx.x, grp = lane >> 4, mi = lane & 15;
    const float* __restrict__ row = X + (mt * 16 + mi) * D_ + kc * 32;
    v8h o0, o1;
    #pragma unroll
    for (int e = 0; e < 8; ++e) {
        o0[e] = (_Float16)row[grp * 16 + e];         // K = first 8 of this group's 16
        o1[e] = (_Float16)row[grp * 16 + 8 + e];     // K = last 8 of this group's 16
    }
    _Float16* __restrict__ dst = apack + ((size_t)(mt * 8 + kc) * 32 + lane) * 16;
    *(v8h*)dst       = o0;
    *(v8h*)(dst + 8) = o1;
}

// ---------------------------------------------------------------------------
// Pre-pack weights: W [256,256] f32 -> f16 B-fragment order.
//   wpack[(((w*16 + nt)*8 + kc)*32 + lane)*16 + e]
// grid: (16 nt, 8 kc, 4 w), block 32
// ---------------------------------------------------------------------------
__global__ __launch_bounds__(32)
void pack_w_kernel(const float* __restrict__ Wq, const float* __restrict__ Wk,
                   const float* __restrict__ Wv, const float* __restrict__ Wo,
                   _Float16* __restrict__ wpack)
{
    const int nt = blockIdx.x, kc = blockIdx.y, w = blockIdx.z;
    const float* __restrict__ W = (w == 0) ? Wq : (w == 1) ? Wk : (w == 2) ? Wv : Wo;
    const int lane = threadIdx.x, grp = lane >> 4, mi = lane & 15;
    const float* __restrict__ col = W + nt * 16 + mi;      // stride D_ along K
    v8h o0, o1;
    #pragma unroll
    for (int e = 0; e < 8; ++e) {
        o0[e] = (_Float16)col[(kc * 32 + grp * 16 + e) * D_];
        o1[e] = (_Float16)col[(kc * 32 + grp * 16 + 8 + e) * D_];
    }
    _Float16* __restrict__ dst =
        wpack + ((size_t)((w * 16 + nt) * 8 + kc) * 32 + lane) * 16;
    *(v8h*)dst       = o0;
    *(v8h*)(dst + 8) = o1;
}

// ---------------------------------------------------------------------------
// Kernel 1: QKV projection from pre-packed fragments.
//   which==0 -> Q (scaled) into qh [B,H,L,DK]
//   which==1 -> K           into kh [B,H,L,DK]
//   which==2 -> V transposed into vt [B,H,DK,L]
// grid: (128 mt, 16 nt, 3), block 32
// ---------------------------------------------------------------------------
__global__ __launch_bounds__(32)
void qkv_project_kernel(const _Float16* __restrict__ apack,
                        const _Float16* __restrict__ wpack,
                        _Float16* __restrict__ qh,
                        _Float16* __restrict__ kh,
                        _Float16* __restrict__ vt)
{
    const int mt = blockIdx.x, nt = blockIdx.y, which = blockIdx.z;
    const int lane = threadIdx.x, grp = lane >> 4;

    const _Float16* __restrict__ abase = apack + ((size_t)mt * 8 * 32 + lane) * 16;
    const _Float16* __restrict__ bbase =
        wpack + ((size_t)(which * 16 + nt) * 8 * 32 + lane) * 16;

    v8f acc = {};
    #pragma unroll
    for (int kc = 0; kc < D_ / 32; ++kc) {
        v16h a = *(const v16h*)(abase + (size_t)kc * 512);   // 2x b128
        v16h b = *(const v16h*)(bbase + (size_t)kc * 512);   // 2x b128
        acc = wmma16(a, b, acc);
    }

    const int ncol = nt * 16 + (lane & 15);
    const int h = ncol >> 5;
    const int d = ncol & 31;
    #pragma unroll
    for (int r = 0; r < 8; ++r) {
        const int tok = mt * 16 + grp * 8 + r;     // global token in [0, B*L)
        const int b   = tok >> 9;
        const int l   = tok & (L_ - 1);
        float v = acc[r];
        if (which == 0) {
            qh[(((b * H_ + h) * L_) + l) * DK_ + d] = (_Float16)(v * SCALE_);
        } else if (which == 1) {
            kh[(((b * H_ + h) * L_) + l) * DK_ + d] = (_Float16)v;
        } else {
            vt[(((b * H_ + h) * DK_) + d) * L_ + l] = (_Float16)v;
        }
    }
}

// ---------------------------------------------------------------------------
// Kernel 2: mixed-score attention, flash-style online softmax.
// One wave per (b, h, 16-row q-tile); 32 keys (2 WMMA tiles) per iteration.
// Mixed-score MLP runs on the packed-f16 VALU pipe (v_pk_fma_f16 /
// v_pk_max) with f32 accumulation of the output layer.
// LDS: only the current 16x32 P tile (1 KB).
// grid: (L/16=32, H, B), block 32
// ---------------------------------------------------------------------------
__global__ __launch_bounds__(32)
void attention_kernel(const float* __restrict__ cost,     // [B, L, L]
                      const float* __restrict__ mix1_w,   // [H, 2, MS]
                      const float* __restrict__ mix1_b,   // [H, MS]
                      const float* __restrict__ mix2_w,   // [H, MS]
                      const float* __restrict__ mix2_b,   // [H]
                      const _Float16* __restrict__ qh,    // [B,H,L,DK]
                      const _Float16* __restrict__ kh,    // [B,H,L,DK]
                      const _Float16* __restrict__ vt,    // [B,H,DK,L]
                      _Float16* __restrict__ atth)        // [B*L, H*DK]
{
    __shared__ _Float16 s_p[16 * 32];    // current P tile, 1 KB

    const int qt   = blockIdx.x;
    const int h    = blockIdx.y;
    const int b    = blockIdx.z;
    const int bh   = b * H_ + h;
    const int lane = threadIdx.x;
    const int grp  = lane >> 4;
    const int mi   = lane & 15;

    // Per-head MLP parameters, packed for v_pk_* (wave-uniform addresses).
    h2 w10p[MS_ / 2], w11p[MS_ / 2], b1p[MS_ / 2];
    float w2v[MS_];
    #pragma unroll
    for (int i = 0; i < MS_ / 2; ++i) {
        w10p[i][0] = (_Float16)mix1_w[(h * 2 + 0) * MS_ + 2 * i];
        w10p[i][1] = (_Float16)mix1_w[(h * 2 + 0) * MS_ + 2 * i + 1];
        w11p[i][0] = (_Float16)mix1_w[(h * 2 + 1) * MS_ + 2 * i];
        w11p[i][1] = (_Float16)mix1_w[(h * 2 + 1) * MS_ + 2 * i + 1];
        b1p[i][0]  = (_Float16)mix1_b[h * MS_ + 2 * i];
        b1p[i][1]  = (_Float16)mix1_b[h * MS_ + 2 * i + 1];
    }
    #pragma unroll
    for (int i = 0; i < MS_; ++i) w2v[i] = mix2_w[h * MS_ + i];
    const float b2v = mix2_b[h];

    // Q A-fragment (pre-scaled by DK^-0.5 at projection time).
    const _Float16* __restrict__ qrow = qh + (bh * L_ + qt * 16 + mi) * DK_;
    const v16h aq = combine(*(const v8h*)(qrow + grp * 16),
                            *(const v8h*)(qrow + grp * 16 + 8));

    float m[8], sum[8];
    #pragma unroll
    for (int r = 0; r < 8; ++r) { m[r] = -1e30f; sum[r] = 0.0f; }
    v8f acc0 = {}, acc1 = {};

    const _Float16* __restrict__ vb0 = vt + (bh * DK_ + mi) * L_;
    const _Float16* __restrict__ vb1 = vt + (bh * DK_ + 16 + mi) * L_;

    for (int kc = 0; kc < L_ / 32; ++kc) {
        float s0[8], s1[8];
        // ---- two 16-key score tiles: WMMA + packed-f16 mixed-score MLP ----
        #pragma unroll
        for (int t = 0; t < 2; ++t) {
            const int kt = kc * 2 + t;
            const _Float16* __restrict__ krow =
                kh + (bh * L_ + kt * 16 + mi) * DK_ + grp * 16;
            if (kt + 2 < L_ / 16)
                __builtin_prefetch(krow + 32 * DK_, 0, 1);   // global_prefetch_b8
            const v16h bk = combine(*(const v8h*)(krow),
                                    *(const v8h*)(krow + 8));
            v8f c = {};
            c = wmma16(aq, bk, c);

            const int col = kt * 16 + mi;
            #pragma unroll
            for (int r = 0; r < 8; ++r) {
                const int rl   = grp * 8 + r;
                const float dv = c[r];
                const float cv = cost[(b * L_ + qt * 16 + rl) * L_ + col];
                const _Float16 dh = (_Float16)dv, ch = (_Float16)cv;
                const h2 dv2 = { dh, dh };
                const h2 cv2 = { ch, ch };
                float sa = b2v, sb = 0.0f;     // f32 output-layer accumulators
                #pragma unroll
                for (int i = 0; i < MS_ / 2; ++i) {
                    h2 a1 = dv2 * w10p[i] + (cv2 * w11p[i] + b1p[i]); // pk_fma
                    a1 = __builtin_elementwise_max(a1, (h2)0);        // pk_max
                    sa = fmaf((float)a1[0], w2v[2 * i], sa);          // fma_mix
                    sb = fmaf((float)a1[1], w2v[2 * i + 1], sb);
                }
                const float s = sa + sb;
                if (t == 0) s0[r] = s; else s1[r] = s;
            }
        }

        // ---- online softmax update + stage P tile to LDS ----
        #pragma unroll
        for (int r = 0; r < 8; ++r) {
            float tm = fmaxf(s0[r], s1[r]);
            #pragma unroll
            for (int off = 1; off < 16; off <<= 1)
                tm = fmaxf(tm, __shfl_xor(tm, off, 16));
            const float nm = fmaxf(m[r], tm);
            const float sc = __expf(m[r] - nm);
            const float p0 = __expf(s0[r] - nm);
            const float p1 = __expf(s1[r] - nm);
            float ps = p0 + p1;
            #pragma unroll
            for (int off = 1; off < 16; off <<= 1)
                ps += __shfl_xor(ps, off, 16);
            sum[r] = sum[r] * sc + ps;
            m[r]   = nm;
            acc0[r] *= sc;
            acc1[r] *= sc;
            const int rl = grp * 8 + r;
            s_p[rl * 32 + mi]      = (_Float16)p0;
            s_p[rl * 32 + 16 + mi] = (_Float16)p1;
        }
        __syncthreads();

        // ---- PV: P(16x32) @ V(32x32) as two N-tiles ----
        const _Float16* __restrict__ prow = &s_p[mi * 32];
        const v16h ap = combine(*(const v8h*)(prow + grp * 16),
                                *(const v8h*)(prow + grp * 16 + 8));
        const _Float16* __restrict__ v0 = vb0 + kc * 32 + grp * 16;
        const _Float16* __restrict__ v1 = vb1 + kc * 32 + grp * 16;
        const v16h bv0 = combine(*(const v8h*)(v0), *(const v8h*)(v0 + 8));
        const v16h bv1 = combine(*(const v8h*)(v1), *(const v8h*)(v1 + 8));
        acc0 = wmma16(ap, bv0, acc0);
        acc1 = wmma16(ap, bv1, acc1);
        __syncthreads();
    }

    // ---- finalize: normalize and store f16 att tile ----
    #pragma unroll
    for (int r = 0; r < 8; ++r) {
        const float rinv = 1.0f / sum[r];
        const int rl  = grp * 8 + r;
        const int tok = b * L_ + qt * 16 + rl;
        atth[tok * (H_ * DK_) + h * DK_ + mi]      = (_Float16)(acc0[r] * rinv);
        atth[tok * (H_ * DK_) + h * DK_ + 16 + mi] = (_Float16)(acc1[r] * rinv);
    }
}

// ---------------------------------------------------------------------------
// Kernel 3: output projection  att[B*L, 256](f16) @ Wo(packed f16) -> f32
// grid: (128, 16), block 32
// ---------------------------------------------------------------------------
__global__ __launch_bounds__(32)
void out_proj_kernel(const _Float16* __restrict__ atth,
                     const _Float16* __restrict__ wpack,
                     float* __restrict__ out)
{
    const int mt   = blockIdx.x;
    const int nt   = blockIdx.y;
    const int lane = threadIdx.x;
    const int grp  = lane >> 4;
    const int mi   = lane & 15;

    const _Float16* __restrict__ arow = atth + (mt * 16 + mi) * (H_ * DK_);
    const _Float16* __restrict__ bbase =
        wpack + ((size_t)((3 * 16 + nt) * 8) * 32 + lane) * 16;   // Wo slot

    v8f acc = {};
    #pragma unroll
    for (int kc = 0; kc < (H_ * DK_) / 32; ++kc) {
        const int kbase = kc * 32;
        const v16h a = combine(*(const v8h*)(arow + kbase + grp * 16),
                               *(const v8h*)(arow + kbase + grp * 16 + 8));
        const v16h bf = *(const v16h*)(bbase + (size_t)kc * 512);
        acc = wmma16(a, bf, acc);
    }
    #pragma unroll
    for (int r = 0; r < 8; ++r)
        out[(mt * 16 + grp * 8 + r) * D_ + nt * 16 + mi] = acc[r];
}

// ---------------------------------------------------------------------------
extern "C" void kernel_launch(void* const* d_in, const int* in_sizes, int n_in,
                              void* d_out, int out_size, void* d_ws, size_t ws_size,
                              hipStream_t stream)
{
    (void)in_sizes; (void)n_in; (void)out_size; (void)ws_size;
    const float* queries = (const float*)d_in[0];   // [B,L,D]
    const float* cost    = (const float*)d_in[1];   // [B,L,L]
    const float* Wq      = (const float*)d_in[2];   // [D, H*DK]
    const float* Wk      = (const float*)d_in[3];
    const float* Wv      = (const float*)d_in[4];
    const float* Wo      = (const float*)d_in[5];   // [H*DK, D]
    const float* mix1_w  = (const float*)d_in[6];   // [H,2,MS]
    const float* mix1_b  = (const float*)d_in[7];   // [H,MS]
    const float* mix2_w  = (const float*)d_in[8];   // [H,MS,1]
    const float* mix2_b  = (const float*)d_in[9];   // [H,1]

    // Workspace layout (f16 elements):
    //   qh/kh/vt/atth : 4 x 524288  (1 MB each)
    //   apack         : 524288      (1 MB)  queries in A-fragment order
    //   wpack         : 262144      (512 KB) Wq|Wk|Wv|Wo in B-fragment order
    const size_t NELT = (size_t)B_ * H_ * L_ * DK_;         // 524288
    _Float16* qh    = (_Float16*)d_ws;
    _Float16* kh    = qh + NELT;
    _Float16* vt    = kh + NELT;
    _Float16* atth  = vt + NELT;
    _Float16* apack = atth + NELT;
    _Float16* wpack = apack + NELT;

    pack_a_kernel<<<dim3((B_ * L_) / 16, D_ / 32), 32, 0, stream>>>(queries, apack);
    pack_w_kernel<<<dim3(D_ / 16, D_ / 32, 4), 32, 0, stream>>>(Wq, Wk, Wv, Wo, wpack);
    qkv_project_kernel<<<dim3((B_ * L_) / 16, (H_ * DK_) / 16, 3), 32, 0, stream>>>(
        apack, wpack, qh, kh, vt);
    attention_kernel<<<dim3(L_ / 16, H_, B_), 32, 0, stream>>>(
        cost, mix1_w, mix1_b, mix2_w, mix2_b, qh, kh, vt, atth);
    out_proj_kernel<<<dim3((B_ * L_) / 16, D_ / 16), 32, 0, stream>>>(
        atth, wpack, (float*)d_out);
}
